// BiLSTMRelPN_37005438222791
// MI455X (gfx1250) — matrix operand order repaired and checked by reading, not compiled, not run on hardware
//
#include <hip/hip_runtime.h>
#include <hip/hip_bf16.h>
#include <math.h>
#include <stdint.h>

// ---------------------------------------------------------------------------
// Types / helpers
// ---------------------------------------------------------------------------
typedef __attribute__((ext_vector_type(16))) __bf16 v16bf;
typedef __attribute__((ext_vector_type(8)))  float  v8f;

struct alignas(16) BF8 { __bf16 v[8]; };

__device__ __forceinline__ __bf16 f2bf(float f) {
    unsigned u = __float_as_uint(f);
    unsigned r = (u + 0x7FFFu + ((u >> 16) & 1u)) >> 16;   // RNE
    unsigned short s = (unsigned short)r;
    __bf16 o;
    __builtin_memcpy(&o, &s, 2);
    return o;
}

// Async global->LDS copy of 16 bytes (one b128), tracked by ASYNCcnt.
// dsaddr = low 32 bits of the generic LDS pointer (LDS aperture offset).
__device__ __forceinline__ void async_copy_b128(const __bf16* gsrc, __bf16* ldst) {
    unsigned           l = (unsigned)(uintptr_t)ldst;
    unsigned long long g = (unsigned long long)(uintptr_t)gsrc;
    asm volatile("global_load_async_to_lds_b128 %0, %1, off"
                 :: "v"(l), "v"(g) : "memory");
}
__device__ __forceinline__ void wait_async0() {
    asm volatile("s_wait_asynccnt 0x0" ::: "memory");
}

// A-matrix fragment (16x32, 16-bit): lane(0-15)=row M, sel=lane>>4 picks
// K chunks {sel*8 .. sel*8+7} (VGPR0-3) and {sel*8+16 .. sel*8+23} (VGPR4-7).
__device__ __forceinline__ v16bf load_fragA(const __bf16* base, int ld,
                                            int row0, int k0, int lane) {
    int l = lane & 15, sel = (lane >> 4) & 1;
    const __bf16* p = base + (size_t)(row0 + l) * ld + k0 + sel * 8;
    BF8 lo = *(const BF8*)p;
    BF8 hi = *(const BF8*)(p + 16);
    v16bf r;
#pragma unroll
    for (int i = 0; i < 8; ++i) { r[i] = lo.v[i]; r[i + 8] = hi.v[i]; }
    return r;
}

// B-matrix fragment (32x16, 16-bit): lane(0-15)=col N, sel=lane>>4 picks
// K chunks {sel*8 .. sel*8+7} and {sel*8+16 .. sel*8+23} (same as A).
__device__ __forceinline__ v16bf load_fragB(const __bf16* base, int ld,
                                            int n0, int k0, int lane) {
    int l = lane & 15, sel = (lane >> 4) & 1;
    const __bf16* p = base + (size_t)(n0 + l) * ld + k0 + sel * 8;
    BF8 lo = *(const BF8*)p;
    BF8 hi = *(const BF8*)(p + 16);
    v16bf r;
#pragma unroll
    for (int i = 0; i < 8; ++i) { r[i] = lo.v[i]; r[i + 8] = hi.v[i]; }
    return r;
}

__device__ __forceinline__ v8f wmma_bf16(v16bf a, v16bf b, v8f c) {
    return __builtin_amdgcn_wmma_f32_16x16x32_bf16(
        /*neg_a=*/false, a, /*neg_b=*/false, b,
        /*c_mod=*/(short)0, c, /*reuse_a=*/false, /*reuse_b=*/false);
}

__device__ __forceinline__ float sigmoidf_(float x) {
    return 1.0f / (1.0f + expf(-x));
}

// ---------------------------------------------------------------------------
// Problem constants
// ---------------------------------------------------------------------------
#define LSEQ 1024
#define BAT  32
#define ISZ  256
#define HSZ  256
#define DMODEL 512           // 2*H
#define NGATE  1024          // 4*H
#define KTOP 3

// ---------------------------------------------------------------------------
// Kernel 0a: f32 -> bf16 conversion of the input sequence
// ---------------------------------------------------------------------------
__global__ void cvt_x_kernel(const float* __restrict__ x,
                             __bf16* __restrict__ out, int n) {
    for (int i = blockIdx.x * blockDim.x + threadIdx.x; i < n;
         i += gridDim.x * blockDim.x)
        out[i] = f2bf(x[i]);
}

// ---------------------------------------------------------------------------
// Kernel 0b: pack per-direction weights as bf16 [W_ih | W_hh] (1024 x 512)
// ---------------------------------------------------------------------------
__global__ void build_w_kernel(const float* __restrict__ wif,
                               const float* __restrict__ whf,
                               const float* __restrict__ wib,
                               const float* __restrict__ whb,
                               __bf16* __restrict__ wc, int n) {
    for (int i = blockIdx.x * blockDim.x + threadIdx.x; i < n;
         i += gridDim.x * blockDim.x) {
        int dir = i >> 19;                 // 1024*512 = 2^19 per direction
        int rem = i & ((1 << 19) - 1);
        int g = rem >> 9;                  // gate-unit row 0..1023
        int k = rem & 511;                 // 0..255 -> W_ih, 256..511 -> W_hh
        const float* wi = dir ? wib : wif;
        const float* wh = dir ? whb : whf;
        float v = (k < 256) ? wi[g * 256 + k] : wh[g * 256 + (k - 256)];
        wc[i] = f2bf(v);
    }
}

// ---------------------------------------------------------------------------
// Kernel 1: persistent BiLSTM. One 1024-thread block per direction.
//   32 waves: wave = (mtile<<4) | hslice. Each wave owns a 16-wide h-slice
//   and computes the four gate tiles {i,f,g,o} for it, so the pointwise
//   update is fully in-lane (C-layout puts i/f/g/o of one (b,h) in one lane).
//   x_t is double-buffered in LDS via async global->LDS b128 copies
//   (ASYNCcnt), overlapping the next step's fetch with this step's WMMAs.
// ---------------------------------------------------------------------------
__global__ __launch_bounds__(1024) void
bilstm_kernel(const __bf16* __restrict__ xb,      // (L,B,I) bf16
              const __bf16* __restrict__ wcomb,   // (2,1024,512) bf16
              const float*  __restrict__ biasf,   // (4H)
              const float*  __restrict__ biasb,   // (4H)
              float*  __restrict__ lstm_out,      // (B,L,2H) f32
              __bf16* __restrict__ lstm_bf)       // (B,L,2H) bf16
{
    __shared__ __bf16 xbuf[2][BAT][ISZ];          // 32 KB: x_t double buffer
    __shared__ __bf16 hbuf[BAT][HSZ];             // 16 KB: current h (bf16)

    const int dir   = blockIdx.x;
    const float* bias = dir ? biasb : biasf;
    const __bf16* w   = wcomb + (size_t)dir * (NGATE * 512);

    const int tid  = threadIdx.x;
    const int wave = tid >> 5;
    const int lane = tid & 31;
    const int mtile = wave >> 4;                  // 0..1 (batch halves)
    const int hbase = (wave & 15) * 16;           // h-slice start
    const int row0  = mtile * 16;
    const int l15 = lane & 15, sel = (lane >> 4) & 1;

    // Preload x_{t0} into buffer 0 (each thread moves one 16B chunk).
    {
        const int t0 = dir ? (LSEQ - 1) : 0;
        async_copy_b128(xb + (size_t)t0 * (BAT * ISZ) + tid * 8,
                        &xbuf[0][0][0] + tid * 8);
    }
    for (int i = tid; i < BAT * HSZ; i += 1024)
        ((__bf16*)hbuf)[i] = f2bf(0.0f);

    float bv[4];
#pragma unroll
    for (int g = 0; g < 4; ++g) bv[g] = bias[g * HSZ + hbase + l15];

    float cst[8];
#pragma unroll
    for (int r = 0; r < 8; ++r) cst[r] = 0.0f;

    wait_async0();
    __syncthreads();

    for (int s = 0; s < LSEQ; ++s) {
        const int t   = dir ? (LSEQ - 1 - s) : s;
        const int cur = s & 1;

        // Kick off async fetch of x_{t+1} into the other buffer. Its previous
        // readers (step s-1 GEMM) are already fenced by the barriers below.
        if (s + 1 < LSEQ) {
            const int tn = dir ? (LSEQ - 2 - s) : (s + 1);
            async_copy_b128(xb + (size_t)tn * (BAT * ISZ) + tid * 8,
                            &xbuf[1 - cur][0][0] + tid * 8);
        }

        v8f acc[4] = {};

        // K = 0..255 : x_t @ W_ih^T  (A from LDS double buffer)
        for (int kk = 0; kk < 8; ++kk) {
            v16bf a = load_fragA(&xbuf[cur][0][0], ISZ, row0, kk * 32, lane);
#pragma unroll
            for (int g = 0; g < 4; ++g) {
                v16bf bb = load_fragB(w, 512, g * HSZ + hbase, kk * 32, lane);
                acc[g] = wmma_bf16(a, bb, acc[g]);
            }
        }
        // K = 256..511 : h_{t-1} @ W_hh^T  (A from LDS)
        for (int kk = 0; kk < 8; ++kk) {
            v16bf a = load_fragA(&hbuf[0][0], HSZ, row0, kk * 32, lane);
#pragma unroll
            for (int g = 0; g < 4; ++g) {
                v16bf bb = load_fragB(w, 512, g * HSZ + hbase, 256 + kk * 32, lane);
                acc[g] = wmma_bf16(a, bb, acc[g]);
            }
        }

        __syncthreads();   // everyone finished reading hbuf (and xbuf[cur])

        // C layout: element r -> (b = row0 + 8*sel + r, h = hbase + l15);
        // gates i/f/g/o for that element are acc[0..3][r] in this lane.
#pragma unroll
        for (int r = 0; r < 8; ++r) {
            const int b  = row0 + sel * 8 + r;
            const int hc = hbase + l15;
            float ig = sigmoidf_(acc[0][r] + bv[0]);
            float fg = sigmoidf_(acc[1][r] + bv[1]);
            float gg = tanhf   (acc[2][r] + bv[2]);
            float og = sigmoidf_(acc[3][r] + bv[3]);
            float c = fg * cst[r] + ig * gg;
            cst[r] = c;
            float h = og * tanhf(c);
            __bf16 hb = f2bf(h);
            hbuf[b][hc] = hb;
            size_t oi = ((size_t)b * LSEQ + t) * DMODEL + dir * HSZ + hc;
            lstm_out[oi] = h;
            lstm_bf[oi]  = hb;
        }

        wait_async0();     // x_{t+1} landed in LDS
        __syncthreads();   // h + next x published before next step's GEMM
    }
}

// ---------------------------------------------------------------------------
// Kernel 2: align = (lstm lstm^T)/sqrt(d) per batch, streamed through LDS in
// 16-row strips; fused top-3 (descending, first-index tie-break) -> coo/vals.
// The 16x512 A-strip is staged into LDS once via async b128 copies.
// grid = (L/16, B), block = 256 threads (8 waves).
// ---------------------------------------------------------------------------
__global__ __launch_bounds__(256) void
align_topk_kernel(const __bf16* __restrict__ lstm_bf,   // (B,L,2H) bf16
                  float* __restrict__ coo,              // (B,2,L*K)
                  float* __restrict__ vals)             // (B,L,K)
{
    __shared__ __bf16 astrip[16][DMODEL];               // 16 KB A-strip
    __shared__ float arow[16][LSEQ];                    // 64 KB strip
    __shared__ float sv[16][16][KTOP];
    __shared__ int   si[16][16][KTOP];

    const int b  = blockIdx.y;
    const int lt = blockIdx.x;
    const int l0 = lt * 16;
    const __bf16* base = lstm_bf + (size_t)b * LSEQ * DMODEL;

    const int wave = threadIdx.x >> 5;
    const int lane = threadIdx.x & 31;
    const int l15 = lane & 15, sel = (lane >> 4) & 1;
    const float scale = 0.044194173824159216f;          // 1/sqrt(512)

    // Stage the 16x512 bf16 A-strip (contiguous rows l0..l0+15) into LDS.
#pragma unroll
    for (int j = 0; j < 4; ++j) {
        const int c = threadIdx.x + j * 256;            // 1024 chunks of 8 elems
        async_copy_b128(base + (size_t)l0 * DMODEL + c * 8,
                        &astrip[0][0] + c * 8);
    }
    wait_async0();
    __syncthreads();

    // 8 waves x 8 n-tiles each = 64 n-tiles covering m = 0..1023
    for (int j = 0; j < 8; ++j) {
        const int nt = wave * 8 + j;
        v8f acc = {};
        for (int kk = 0; kk < 16; ++kk) {
            v16bf a  = load_fragA(&astrip[0][0], DMODEL, 0, kk * 32, lane);
            v16bf bb = load_fragB(base, DMODEL, nt * 16, kk * 32, lane);
            acc = wmma_bf16(a, bb, acc);
        }
        const int n = nt * 16 + l15;
#pragma unroll
        for (int r = 0; r < 8; ++r)
            arow[sel * 8 + r][n] = acc[r] * scale;
    }
    __syncthreads();

    // Stage 1: 16 threads per row scan 64 columns each for local top-3.
    {
        const int row = threadIdx.x >> 4;
        const int seg = threadIdx.x & 15;
        float v0 = -3.402823e38f, v1 = v0, v2 = v0;
        int   i0 = 0, i1 = 0, i2 = 0;
        for (int j = 0; j < 64; ++j) {
            int col = seg * 64 + j;
            float v = arow[row][col];
            if (v > v0)      { v2=v1; i2=i1; v1=v0; i1=i0; v0=v; i0=col; }
            else if (v > v1) { v2=v1; i2=i1; v1=v;  i1=col; }
            else if (v > v2) { v2=v;  i2=col; }
        }
        sv[row][seg][0]=v0; sv[row][seg][1]=v1; sv[row][seg][2]=v2;
        si[row][seg][0]=i0; si[row][seg][1]=i1; si[row][seg][2]=i2;
    }
    __syncthreads();

    // Stage 2: one thread per row merges 16 candidate triples (ascending
    // segment order + strict '>' preserves first-index tie-breaking).
    if (threadIdx.x < 16) {
        const int row = threadIdx.x;
        float v0 = -3.402823e38f, v1 = v0, v2 = v0;
        int   i0 = 0, i1 = 0, i2 = 0;
        for (int seg = 0; seg < 16; ++seg)
#pragma unroll
            for (int c = 0; c < KTOP; ++c) {
                float v = sv[row][seg][c];
                int  ix = si[row][seg][c];
                if (v > v0)      { v2=v1; i2=i1; v1=v0; i1=i0; v0=v; i0=ix; }
                else if (v > v1) { v2=v1; i2=i1; v1=v;  i1=ix; }
                else if (v > v2) { v2=v;  i2=ix; }
            }
        const int l = l0 + row;
        float*  vo = vals + ((size_t)b * LSEQ + l) * KTOP;
        float*  c0 = coo + (size_t)b * 2 * (LSEQ * KTOP) + l * KTOP;
        float*  c1 = c0 + (LSEQ * KTOP);
        vo[0] = v0; vo[1] = v1; vo[2] = v2;
        c0[0] = (float)i0; c0[1] = (float)i1; c0[2] = (float)i2;
        c1[0] = (float)l;  c1[1] = (float)l;  c1[2] = (float)l;
    }
}

// ---------------------------------------------------------------------------
// Launch
// ---------------------------------------------------------------------------
extern "C" void kernel_launch(void* const* d_in, const int* in_sizes, int n_in,
                              void* d_out, int out_size, void* d_ws, size_t ws_size,
                              hipStream_t stream) {
    (void)in_sizes; (void)n_in; (void)out_size; (void)ws_size;

    const float* x      = (const float*)d_in[0];   // (L,B,I)
    const float* w_ih_f = (const float*)d_in[1];
    const float* w_hh_f = (const float*)d_in[2];
    const float* b_f    = (const float*)d_in[3];
    const float* w_ih_b = (const float*)d_in[4];
    const float* w_hh_b = (const float*)d_in[5];
    const float* b_b    = (const float*)d_in[6];

    // Workspace carve-up (bf16 elements): x (8M), packed W (1M), lstm copy (16M)
    __bf16* xb  = (__bf16*)d_ws;
    __bf16* wc  = xb + (size_t)LSEQ * BAT * ISZ;          // + 8,388,608
    __bf16* lbf = wc + (size_t)2 * NGATE * 512;           // + 1,048,576

    // Output layout: coo (B*2*L*K) | vals (B*L*K) | lstm_out (B*L*2H)
    float* out  = (float*)d_out;
    float* coo  = out;
    float* vals = out + (size_t)BAT * 2 * LSEQ * KTOP;    // +196608
    float* lstm = vals + (size_t)BAT * LSEQ * KTOP;       // +98304

    cvt_x_kernel<<<dim3(4096), dim3(256), 0, stream>>>(
        x, xb, LSEQ * BAT * ISZ);
    build_w_kernel<<<dim3(1024), dim3(256), 0, stream>>>(
        w_ih_f, w_hh_f, w_ih_b, w_hh_b, wc, 2 * NGATE * 512);
    bilstm_kernel<<<dim3(2), dim3(1024), 0, stream>>>(
        xb, wc, b_f, b_b, lstm, lbf);
    align_topk_kernel<<<dim3(LSEQ / 16, BAT), dim3(256), 0, stream>>>(
        lbf, coo, vals);
}
